// MyGnn_68693706932387
// MI455X (gfx1250) — matrix operand order repaired and checked by reading, not compiled, not run on hardware
//
#include <hip/hip_runtime.h>
#include <hip/hip_bf16.h>
#include <math.h>

#define NN 50000L
#define NE 800000L

typedef __attribute__((ext_vector_type(16))) __bf16        v16bf;
typedef __attribute__((ext_vector_type(8)))  float         v8f;
typedef __attribute__((ext_vector_type(4)))  unsigned int  v4u;
typedef __attribute__((ext_vector_type(8)))  unsigned int  v8u;

__device__ __forceinline__ unsigned short f2bf(float f) {
    unsigned u = __float_as_uint(f);
    u += 0x7FFFu + ((u >> 16) & 1u);           // round-to-nearest-even
    return (unsigned short)(u >> 16);
}

__device__ __forceinline__ v16bf load_frag(const unsigned short* p0,
                                           const unsigned short* p1) {
    v4u a = *(const v4u*)p0;                    // 16B
    v4u b = *(const v4u*)p1;                    // 16B
    v8u u = __builtin_shufflevector(a, b, 0, 1, 2, 3, 4, 5, 6, 7);
    return __builtin_bit_cast(v16bf, u);
}

__device__ __forceinline__ void atomicMinF(float* a, float v) {
    if (v >= 0.f) atomicMin((int*)a, __float_as_int(v));
    else          atomicMax((unsigned int*)a, __float_as_uint(v));
}
__device__ __forceinline__ void atomicMaxF(float* a, float v) {
    if (v >= 0.f) atomicMax((int*)a, __float_as_int(v));
    else          atomicMin((unsigned int*)a, __float_as_uint(v));
}

// ---------------------------------------------------------------- utility
__global__ void k_fill_minmax(float* mm) {
    if (threadIdx.x == 0 && blockIdx.x == 0) {
        mm[0] =  __builtin_inff();  mm[1] = -__builtin_inff();   // h min/max
        mm[2] =  __builtin_inff();  mm[3] = -__builtin_inff();   // m min/max
    }
}

__global__ void k_deg(const int* __restrict__ dst, float* __restrict__ deg, int E) {
    int e = blockIdx.x * blockDim.x + threadIdx.x;
    if (e < E) atomicAdd(&deg[dst[e]], 1.0f);
}

// agg[dst[e], f] += x[src[e], f]   (mean aggregation, scatter side)
__global__ void k_scatter(const float* __restrict__ x, const int* __restrict__ src,
                          const int* __restrict__ dst, float* __restrict__ agg,
                          long E, int F) {
    long total  = E * (long)F;
    long stride = (long)gridDim.x * blockDim.x;
    for (long i = blockIdx.x * (long)blockDim.x + threadIdx.x; i < total; i += stride) {
        long e = i / F;
        int  f = (int)(i - e * F);
        atomicAdd(&agg[(long)dst[e] * F + f], x[(long)src[e] * F + f]);
    }
}

// f32 [R,C] -> bf16 [R,Cp] zero-padded
__global__ void k_convert_pad(const float* __restrict__ s, unsigned short* __restrict__ d,
                              long R, int C, int Cp) {
    long total = R * (long)Cp;
    long i = blockIdx.x * (long)blockDim.x + threadIdx.x;
    if (i >= total) return;
    long r = i / Cp;
    int  c = (int)(i - r * Cp);
    d[i] = (c < C) ? f2bf(s[r * C + c]) : (unsigned short)0;
}

// mean = agg / max(deg,1), stored bf16 padded
__global__ void k_mean_bf(const float* __restrict__ agg, const float* __restrict__ deg,
                          unsigned short* __restrict__ d, long R, int C, int Cp) {
    long total = R * (long)Cp;
    long i = blockIdx.x * (long)blockDim.x + threadIdx.x;
    if (i >= total) return;
    long r = i / Cp;
    int  c = (int)(i - r * Cp);
    float v = (c < C) ? agg[r * C + c] / fmaxf(deg[r], 1.0f) : 0.0f;
    d[i] = f2bf(v);
}

// ---------------------------------------------------------------- WMMA GEMM
// C[M,N] = relu?( A[M,K]@B[N,K]^T + A2[M,K2]@B2[N,K2]^T + bias[N] )
// A,A2: bf16 row-major, K multiple of 32, rows 64B-aligned.
// B,B2: bf16 row-major [N,K] (== W as given: out = x @ W.T).
// One 16x16 tile per wave32; 8 waves per block.
__global__ void k_gemm_wmma(const unsigned short* __restrict__ A,
                            const unsigned short* __restrict__ B,
                            const unsigned short* __restrict__ A2,
                            const unsigned short* __restrict__ B2,
                            const float* __restrict__ bias,
                            float* __restrict__ C,
                            int M, int N, int K, int K2, int relu) {
    int tile = blockIdx.x * (blockDim.x >> 5) + (threadIdx.x >> 5);
    int tnc  = N >> 4;
    int tm   = tile / tnc;
    int tn   = tile - tm * tnc;
    if (tm >= (M >> 4)) return;                  // uniform per-wave exit (EXEC all-1)

    int l    = threadIdx.x & 31;
    int half = l >> 4;
    int mrow = (tm << 4) + (l & 15);             // A row for this lane
    int ncol = (tn << 4) + (l & 15);             // B row (= output column)

    v8f acc = {0.f, 0.f, 0.f, 0.f, 0.f, 0.f, 0.f, 0.f};
    const unsigned short* Ar = A + (long)mrow * K;
    const unsigned short* Br = B + (long)ncol * K;
    int aoff = half * 8;                         // A: K = k0 + half*8 (+16)
    int boff = half * 8;                         // B: K = k0 + half*8 (+16)
    for (int k0 = 0; k0 < K; k0 += 32) {
        v16bf av = load_frag(Ar + k0 + aoff, Ar + k0 + aoff + 16);
        v16bf bv = load_frag(Br + k0 + boff, Br + k0 + boff + 16);
        acc = __builtin_amdgcn_wmma_f32_16x16x32_bf16(false, av, false, bv,
                                                      (short)0, acc, false, false);
    }
    if (A2 != nullptr) {
        const unsigned short* Ar2 = A2 + (long)mrow * K2;
        const unsigned short* Br2 = B2 + (long)ncol * K2;
        for (int k0 = 0; k0 < K2; k0 += 32) {
            v16bf av = load_frag(Ar2 + k0 + aoff, Ar2 + k0 + aoff + 16);
            v16bf bv = load_frag(Br2 + k0 + boff, Br2 + k0 + boff + 16);
            acc = __builtin_amdgcn_wmma_f32_16x16x32_bf16(false, av, false, bv,
                                                          (short)0, acc, false, false);
        }
    }
    float bvadd = bias[ncol];
#pragma unroll
    for (int r = 0; r < 8; ++r) {                // C layout: row = half*8+r, col = l&15
        int row = (tm << 4) + half * 8 + r;
        float v = acc[r] + bvadd;
        if (relu) v = fmaxf(v, 0.f);
        C[(long)row * N + (tn << 4) + (l & 15)] = v;
    }
}

// ---------------------------------------------------------------- map encoder
// in [15,15,15], w [256,3,5,5] -> out [256,13,11,11], relu
__global__ void k_conv1(const float* __restrict__ in, const float* __restrict__ w,
                        const float* __restrict__ b, float* __restrict__ out) {
    int i = blockIdx.x * blockDim.x + threadIdx.x;
    if (i >= 256 * 13 * 11 * 11) return;
    int ww = i % 11;  int t = i / 11;
    int hh = t % 11;  t /= 11;
    int dd = t % 13;  int c = t / 13;
    float acc = b[c];
    const float* wc = w + c * 75;
    for (int kd = 0; kd < 3; ++kd)
        for (int kh = 0; kh < 5; ++kh)
            for (int kw = 0; kw < 5; ++kw)
                acc += in[(dd + kd) * 225 + (hh + kh) * 15 + (ww + kw)]
                     * wc[kd * 25 + kh * 5 + kw];
    out[i] = fmaxf(acc, 0.f);
}

// m1 [256,13,11,11], k2 [128,256,3,3,3] -> out [128,11,9,9], relu
__global__ void k_conv2(const float* __restrict__ m1, const float* __restrict__ k2,
                        const float* __restrict__ b, float* __restrict__ out) {
    int i = blockIdx.x * blockDim.x + threadIdx.x;
    if (i >= 128 * 11 * 9 * 9) return;
    int ww = i % 9;  int t = i / 9;
    int hh = t % 9;  t /= 9;
    int dd = t % 11; int c = t / 11;
    float acc = b[c];
    for (int ci = 0; ci < 256; ++ci) {
        const float* mp = m1 + ci * 1573;
        const float* kp = k2 + ((long)c * 256 + ci) * 27;
        for (int kd = 0; kd < 3; ++kd)
            for (int kh = 0; kh < 3; ++kh)
                for (int kw = 0; kw < 3; ++kw)
                    acc += mp[(dd + kd) * 121 + (hh + kh) * 11 + (ww + kw)]
                         * kp[kd * 9 + kh * 3 + kw];
    }
    out[i] = fmaxf(acc, 0.f);
}

// m[o] = Wm[o,:] . v + bm[o], 64 rows of 114048; one block per row
__global__ void k_matvec(const float* __restrict__ Wm, const float* __restrict__ v,
                         const float* __restrict__ bm, float* __restrict__ out) {
    __shared__ float s[256];
    int o = blockIdx.x;
    const float* row = Wm + (long)o * 114048;
    float acc = 0.f;
    for (int i = threadIdx.x; i < 114048; i += 256) acc += row[i] * v[i];
    s[threadIdx.x] = acc;
    __syncthreads();
    for (int st = 128; st > 0; st >>= 1) {
        if (threadIdx.x < st) s[threadIdx.x] += s[threadIdx.x + st];
        __syncthreads();
    }
    if (threadIdx.x == 0) out[o] = s[0] + bm[o];
}

// ---------------------------------------------------------------- normalization
__global__ void k_minmax(const float* __restrict__ x, long n, float* __restrict__ mm) {
    __shared__ float smn[256], smx[256];
    float mn = __builtin_inff(), mx = -__builtin_inff();
    long stride = (long)gridDim.x * blockDim.x;
    for (long i = blockIdx.x * (long)blockDim.x + threadIdx.x; i < n; i += stride) {
        float v = x[i];
        mn = fminf(mn, v);
        mx = fmaxf(mx, v);
    }
    smn[threadIdx.x] = mn; smx[threadIdx.x] = mx;
    __syncthreads();
    for (int st = 128; st > 0; st >>= 1) {
        if (threadIdx.x < st) {
            smn[threadIdx.x] = fminf(smn[threadIdx.x], smn[threadIdx.x + st]);
            smx[threadIdx.x] = fmaxf(smx[threadIdx.x], smx[threadIdx.x + st]);
        }
        __syncthreads();
    }
    if (threadIdx.x == 0) { atomicMinF(&mm[0], smn[0]); atomicMaxF(&mm[1], smx[0]); }
}

__global__ void k_minmax_small(const float* __restrict__ v, int n, float* __restrict__ mm) {
    if (threadIdx.x == 0 && blockIdx.x == 0) {
        float mn = v[0], mx = v[0];
        for (int i = 1; i < n; ++i) { mn = fminf(mn, v[i]); mx = fmaxf(mx, v[i]); }
        mm[2] = mn; mm[3] = mx;
    }
}

__global__ void k_mnorm(const float* __restrict__ mvec, const float* __restrict__ mm,
                        float* __restrict__ mnorm) {
    int i = threadIdx.x;
    if (i < 64) mnorm[i] = 0.65f * (mvec[i] - mm[2]) / (mm[3] - mm[2]);
}

// xhead = 0.35*norm(h3) + 0.65*norm(m)  (broadcast), stored bf16 [NN,64]
__global__ void k_headx(const float* __restrict__ h3, const float* __restrict__ mm,
                        const float* __restrict__ mnorm, unsigned short* __restrict__ xb,
                        long n) {
    long i = blockIdx.x * (long)blockDim.x + threadIdx.x;
    if (i >= n) return;
    float inv = 1.0f / (mm[1] - mm[0]);
    float v = 0.35f * (h3[i] - mm[0]) * inv + mnorm[i & 63];
    xb[i] = f2bf(v);
}

// per node: z = W2 @ y1_row + b2 (3x64), softmax, write f32[N,3]
__global__ void k_head2(const float* __restrict__ y1, const float* __restrict__ W2,
                        const float* __restrict__ b2, float* __restrict__ out, int Nn) {
    int n = blockIdx.x * blockDim.x + threadIdx.x;
    if (n >= Nn) return;
    const float* xr = y1 + (long)n * 64;
    float z[3];
#pragma unroll
    for (int j = 0; j < 3; ++j) {
        float acc = b2[j];
        const float* wr = W2 + j * 64;
        for (int k = 0; k < 64; ++k) acc += xr[k] * wr[k];
        z[j] = acc;
    }
    float m = fmaxf(z[0], fmaxf(z[1], z[2]));
    float e0 = __expf(z[0] - m), e1 = __expf(z[1] - m), e2 = __expf(z[2] - m);
    float inv = 1.0f / (e0 + e1 + e2);
    out[n * 3 + 0] = e0 * inv;
    out[n * 3 + 1] = e1 * inv;
    out[n * 3 + 2] = e2 * inv;
}

// ---------------------------------------------------------------- launcher
extern "C" void kernel_launch(void* const* d_in, const int* in_sizes, int n_in,
                              void* d_out, int out_size, void* d_ws, size_t ws_size,
                              hipStream_t stream) {
    const float* node = (const float*)d_in[0];
    const int*   edge = (const int*)d_in[1];
    const int*   srcv = edge;
    const int*   dstv = edge + NE;
    const float* mapd = (const float*)d_in[2];
    const float* Wl1 = (const float*)d_in[3];  const float* bl1 = (const float*)d_in[4];
    const float* Wr1 = (const float*)d_in[5];
    const float* Wl2 = (const float*)d_in[6];  const float* bl2 = (const float*)d_in[7];
    const float* Wr2 = (const float*)d_in[8];
    const float* Wl3 = (const float*)d_in[9];  const float* bl3 = (const float*)d_in[10];
    const float* Wr3 = (const float*)d_in[11];
    const float* K1  = (const float*)d_in[12]; const float* cb1 = (const float*)d_in[13];
    const float* K2  = (const float*)d_in[14]; const float* cb2 = (const float*)d_in[15];
    const float* Wm  = (const float*)d_in[16]; const float* bm  = (const float*)d_in[17];
    const float* W1  = (const float*)d_in[18]; const float* b1  = (const float*)d_in[19];
    const float* W2  = (const float*)d_in[20]; const float* b2  = (const float*)d_in[21];
    float* out = (float*)d_out;

    char* ws = (char*)d_ws;
    size_t off = 0;
    auto alloc = [&](size_t bytes) -> char* {
        char* p = ws + off;
        off = (off + bytes + 255) & ~(size_t)255;
        return p;
    };

    float*          deg    = (float*)alloc(NN * 4);
    float*          agg1   = (float*)alloc(NN * 21 * 4);
    unsigned short* xin    = (unsigned short*)alloc(NN * 32 * 2);
    unsigned short* mean1  = (unsigned short*)alloc(NN * 32 * 2);
    unsigned short* wl1b   = (unsigned short*)alloc(256 * 32 * 2);
    unsigned short* wr1b   = (unsigned short*)alloc(256 * 32 * 2);
    unsigned short* wl2b   = (unsigned short*)alloc(128 * 256 * 2);
    unsigned short* wr2b   = (unsigned short*)alloc(128 * 256 * 2);
    unsigned short* wl3b   = (unsigned short*)alloc(64 * 128 * 2);
    unsigned short* wr3b   = (unsigned short*)alloc(64 * 128 * 2);
    unsigned short* w1b    = (unsigned short*)alloc(64 * 64 * 2);
    float*          m1     = (float*)alloc(256 * 1573 * 4);
    float*          m2     = (float*)alloc(114048 * 4);
    float*          mvec   = (float*)alloc(64 * 4);
    float*          mnorm  = (float*)alloc(64 * 4);
    float*          minmax = (float*)alloc(4 * 4);
    unsigned short* mean2  = (unsigned short*)alloc(NN * 256 * 2);
    unsigned short* mean3  = (unsigned short*)alloc(NN * 128 * 2);
    char*           regBF  = alloc(NN * 256 * 2);   // h1_bf, later xhead+y1
    char*           regA   = alloc(NN * 256 * 4);   // h1, later h2 + h2_bf
    char*           regB   = alloc(NN * 256 * 4);   // agg2, later agg3 + h3

    unsigned short* h1bf  = (unsigned short*)regBF;
    unsigned short* xhead = (unsigned short*)regBF;                  // reuse after h1bf dead
    float*          y1    = (float*)(regBF + NN * 64 * 2);
    float*          h1    = (float*)regA;
    float*          h2    = (float*)regA;                            // reuse after h1 dead
    unsigned short* h2bf  = (unsigned short*)(regA + NN * 128 * 4);
    float*          agg2  = (float*)regB;
    float*          agg3  = (float*)regB;                            // reuse after agg2 dead
    float*          h3    = (float*)(regB + NN * 128 * 4);

    const int T = 256;
    auto cdiv = [](long a, long b) -> int { return (int)((a + b - 1) / b); };

    // ---- init
    hipMemsetAsync(deg, 0, NN * 4, stream);
    hipMemsetAsync(agg1, 0, NN * 21 * 4, stream);
    k_fill_minmax<<<1, 32, 0, stream>>>(minmax);

    // ---- weight / input staging to bf16
    k_convert_pad<<<cdiv(NN * 32, T), T, 0, stream>>>(node, xin, NN, 21, 32);
    k_convert_pad<<<cdiv(256 * 32, T), T, 0, stream>>>(Wl1, wl1b, 256, 21, 32);
    k_convert_pad<<<cdiv(256 * 32, T), T, 0, stream>>>(Wr1, wr1b, 256, 21, 32);
    k_convert_pad<<<cdiv(128 * 256, T), T, 0, stream>>>(Wl2, wl2b, 128, 256, 256);
    k_convert_pad<<<cdiv(128 * 256, T), T, 0, stream>>>(Wr2, wr2b, 128, 256, 256);
    k_convert_pad<<<cdiv(64 * 128, T), T, 0, stream>>>(Wl3, wl3b, 64, 128, 128);
    k_convert_pad<<<cdiv(64 * 128, T), T, 0, stream>>>(Wr3, wr3b, 64, 128, 128);
    k_convert_pad<<<cdiv(64 * 64, T), T, 0, stream>>>(W1, w1b, 64, 64, 64);

    // ---- degrees (shared by all layers)
    k_deg<<<cdiv(NE, T), T, 0, stream>>>(dstv, deg, (int)NE);

    // ---- SAGE layer 1: 21 -> 256
    k_scatter<<<8192, T, 0, stream>>>(node, srcv, dstv, agg1, NE, 21);
    k_mean_bf<<<cdiv(NN * 32, T), T, 0, stream>>>(agg1, deg, mean1, NN, 21, 32);
    k_gemm_wmma<<<cdiv((NN / 16) * (256 / 16), 8), T, 0, stream>>>(
        mean1, wl1b, xin, wr1b, bl1, h1, (int)NN, 256, 32, 32, 1);

    // ---- SAGE layer 2: 256 -> 128
    hipMemsetAsync(agg2, 0, NN * 256 * 4, stream);
    k_scatter<<<16384, T, 0, stream>>>(h1, srcv, dstv, agg2, NE, 256);
    k_convert_pad<<<cdiv(NN * 256, T), T, 0, stream>>>(h1, h1bf, NN, 256, 256);
    k_mean_bf<<<cdiv(NN * 256, T), T, 0, stream>>>(agg2, deg, mean2, NN, 256, 256);
    k_gemm_wmma<<<cdiv((NN / 16) * (128 / 16), 8), T, 0, stream>>>(
        mean2, wl2b, h1bf, wr2b, bl2, h2, (int)NN, 128, 256, 256, 1);

    // ---- SAGE layer 3: 128 -> 64
    k_convert_pad<<<cdiv(NN * 128, T), T, 0, stream>>>(h2, h2bf, NN, 128, 128);
    hipMemsetAsync(agg3, 0, NN * 128 * 4, stream);
    k_scatter<<<16384, T, 0, stream>>>(h2, srcv, dstv, agg3, NE, 128);
    k_mean_bf<<<cdiv(NN * 128, T), T, 0, stream>>>(agg3, deg, mean3, NN, 128, 128);
    k_gemm_wmma<<<cdiv((NN / 16) * (64 / 16), 8), T, 0, stream>>>(
        mean3, wl3b, h2bf, wr3b, bl3, h3, (int)NN, 64, 128, 128, 0);

    // ---- map encoder
    k_conv1<<<cdiv(256 * 13 * 11 * 11, T), T, 0, stream>>>(mapd, K1, cb1, m1);
    k_conv2<<<cdiv(128 * 11 * 9 * 9, T), T, 0, stream>>>(m1, K2, cb2, m2);
    k_matvec<<<64, T, 0, stream>>>(Wm, m2, bm, mvec);

    // ---- normalize + combine
    k_minmax_small<<<1, 32, 0, stream>>>(mvec, 64, minmax);
    k_minmax<<<4096, T, 0, stream>>>(h3, NN * 64, minmax);
    k_mnorm<<<1, 64, 0, stream>>>(mvec, minmax, mnorm);
    k_headx<<<cdiv(NN * 64, T), T, 0, stream>>>(h3, minmax, mnorm, xhead, NN * 64);

    // ---- head
    k_gemm_wmma<<<cdiv((NN / 16) * (64 / 16), 8), T, 0, stream>>>(
        xhead, w1b, nullptr, nullptr, b1, y1, (int)NN, 64, 64, 0, 1);
    k_head2<<<cdiv(NN, T), T, 0, stream>>>(y1, W2, b2, out, (int)NN);

    (void)in_sizes; (void)n_in; (void)out_size; (void)ws_size;
}
